// SpVecsUnet_63531156242536
// MI455X (gfx1250) — matrix operand order repaired and checked by reading, not compiled, not run on hardware
//
#include <hip/hip_runtime.h>

typedef _Float16 half8 __attribute__((ext_vector_type(8)));
typedef _Float16 v16h  __attribute__((ext_vector_type(16)));
typedef float    v8f   __attribute__((ext_vector_type(8)));
typedef float    f4    __attribute__((ext_vector_type(4)));

#define HH   512
#define WW   512
#define PADW 514
#define NSP  1024
#define PPIX (HH*WW)

// ---------------------------------------------------------------------------
// zero scratch (16B vector stores, grid-stride)
// ---------------------------------------------------------------------------
__global__ void zero16_kernel(f4* __restrict__ p, size_t n16) {
    size_t i = (size_t)blockIdx.x * blockDim.x + threadIdx.x;
    size_t stride = (size_t)gridDim.x * blockDim.x;
    f4 z = {0.f, 0.f, 0.f, 0.f};
    for (; i < n16; i += stride) p[i] = z;
}

// ---------------------------------------------------------------------------
// conv1: 1 -> 64 channels, 3x3 SAME, ReLU. Direct conv (K=9 too small for
// WMMA; bandwidth bound). Output: padded NHWC f16.
// ---------------------------------------------------------------------------
__global__ __launch_bounds__(256)
void conv1_relu_kernel(const float* __restrict__ raw,
                       const float* __restrict__ w1,
                       const float* __restrict__ b1,
                       _Float16* __restrict__ fout) {
    __shared__ float sw[64 * 9];
    const int tid = threadIdx.x;
    for (int i = tid; i < 64 * 9; i += 256) sw[i] = w1[i];
    __syncthreads();

    const int p  = blockIdx.x * 256 + tid;   // one thread per pixel
    const int iy = p >> 9;
    const int ix = p & 511;

    float v[9];
#pragma unroll
    for (int ty = 0; ty < 3; ++ty)
#pragma unroll
        for (int tx = 0; tx < 3; ++tx) {
            int yy = iy + ty - 1, xx = ix + tx - 1;
            v[ty * 3 + tx] = (yy >= 0 && yy < HH && xx >= 0 && xx < WW)
                                 ? raw[yy * WW + xx] : 0.f;
        }

    _Float16 h[64];
#pragma unroll 4
    for (int o = 0; o < 64; ++o) {
        float s = b1[o];
#pragma unroll
        for (int t = 0; t < 9; ++t) s += sw[o * 9 + t] * v[t];
        h[o] = (_Float16)(s > 0.f ? s : 0.f);
    }

    _Float16* dst = fout + ((size_t)(iy + 1) * PADW + (ix + 1)) * 64;
#pragma unroll
    for (int o = 0; o < 64; o += 8) *(half8*)(dst + o) = *(const half8*)(h + o);
}

// ---------------------------------------------------------------------------
// Pack conv weights into per-lane WMMA B-fragment order.
// B 32x16 f16 layout (wave32): lane = N column (lanes 0-15), element e -> K={0..7,16..23};
// lanes 16-31: same N, element e -> K={8..15,24..31}.  K-step kstep covers
// tap = kstep>>1, input channels c0=(kstep&1)*32 .. c0+31.
// ---------------------------------------------------------------------------
__global__ void pack_w2_kernel(const float* __restrict__ w, _Float16* __restrict__ bp) {
    int idx = blockIdx.x * blockDim.x + threadIdx.x;   // [0, 18*4*32)
    if (idx >= 18 * 4 * 32) return;
    const int lane  = idx & 31;
    const int nt    = (idx >> 5) & 3;
    const int kstep = idx >> 7;
    const int n     = (lane & 15) + nt * 16;
    const int tap   = kstep >> 1;
    const int cbase = ((kstep & 1) << 5) + ((lane >= 16) ? 8 : 0);
    _Float16* dst = bp + (size_t)idx * 16;
#pragma unroll
    for (int e = 0; e < 16; ++e)
        dst[e] = (_Float16)w[((size_t)n * 64 + (cbase + (e < 8 ? e : e + 8))) * 9 + tap];
}

__global__ void pack_w3_kernel(const float* __restrict__ w, _Float16* __restrict__ bp) {
    int idx = blockIdx.x * blockDim.x + threadIdx.x;   // [0, 18*32)
    if (idx >= 18 * 32) return;
    const int lane  = idx & 31;
    const int kstep = idx >> 5;
    const int n     = lane & 15;                        // 10..15 -> zero pad
    const int tap   = kstep >> 1;
    const int cbase = ((kstep & 1) << 5) + ((lane >= 16) ? 8 : 0);
    _Float16* dst = bp + (size_t)idx * 16;
#pragma unroll
    for (int e = 0; e < 16; ++e)
        dst[e] = (n < 10) ? (_Float16)w[((size_t)n * 64 + (cbase + (e < 8 ? e : e + 8))) * 9 + tap]
                          : (_Float16)0.f;
}

// ---------------------------------------------------------------------------
// conv2: 64 -> 64, 3x3 SAME, ReLU, implicit GEMM on v_wmma_f32_16x16x32_f16.
// One wave: 64 pixels (4 M-tiles along x) x 64 out channels (4 N-tiles),
// K = 576 fully unrolled in 18 steps of 32.  All loads/stores are base +
// compile-time immediate offset (no address VALU in the hot loop).
// A layout per lane: lanes 0-15 channels {c0..+7, c0+16..+23},
// lanes 16-31 {c0+8..+15, c0+24..+31} -> laneHi folded into the base.
// ---------------------------------------------------------------------------
__global__ __launch_bounds__(256)
void conv2_wmma_kernel(const _Float16* __restrict__ fin,
                       const _Float16* __restrict__ bp,
                       const float* __restrict__ bias,
                       _Float16* __restrict__ fout) {
    const int wave   = (blockIdx.x * blockDim.x + threadIdx.x) >> 5;
    const int lane   = threadIdx.x & 31;
    const int laneHi = lane >> 4;
    const int m      = lane & 15;
    const int iy     = wave >> 3;          // 0..511
    const int x0     = (wave & 7) << 6;    // 0..448 step 64

    const _Float16* baseA[4];
#pragma unroll
    for (int t = 0; t < 4; ++t)
        baseA[t] = fin + ((size_t)iy * PADW + (x0 + t * 16 + m)) * 64
                       + (laneHi ? 8 : 0);
    const _Float16* baseB = bp + (size_t)lane * 16;

    v8f acc[4][4];
#pragma unroll
    for (int t = 0; t < 4; ++t)
#pragma unroll
        for (int nt = 0; nt < 4; ++nt) {
            float b = bias[nt * 16 + m];
#pragma unroll
            for (int r = 0; r < 8; ++r) acc[t][nt][r] = b;
        }

#pragma unroll
    for (int kstep = 0; kstep < 18; ++kstep) {
        const int tap  = kstep >> 1;
        const int ty   = tap / 3, tx = tap % 3;
        const int aoff = (ty * PADW + tx) * 64 + ((kstep & 1) << 5);  // halfs, imm

        v16h a[4];
#pragma unroll
        for (int t = 0; t < 4; ++t) {
            half8 lo = *(const half8*)(baseA[t] + aoff);
            half8 hi = *(const half8*)(baseA[t] + aoff + 16);
#pragma unroll
            for (int e = 0; e < 8; ++e) { a[t][e] = lo[e]; a[t][e + 8] = hi[e]; }
        }
#pragma unroll
        for (int nt = 0; nt < 4; ++nt) {
            const v16h b = *(const v16h*)(baseB + (size_t)(kstep * 4 + nt) * 512);
#pragma unroll
            for (int t = 0; t < 4; ++t)
                acc[t][nt] = __builtin_amdgcn_wmma_f32_16x16x32_f16(
                    false, a[t], false, b, (short)0, acc[t][nt], false, false);
        }
    }

    // ReLU + store f16 (D layout: lane = N col, VGPR r = M row, +8 lanes 16-31)
    _Float16* obase = fout + ((size_t)(iy + 1) * PADW + (x0 + 1 + 8 * laneHi)) * 64 + m;
#pragma unroll
    for (int t = 0; t < 4; ++t)
#pragma unroll
        for (int nt = 0; nt < 4; ++nt)
#pragma unroll
            for (int r = 0; r < 8; ++r) {
                float v = acc[t][nt][r];
                v = v > 0.f ? v : 0.f;
                obase[(size_t)(t * 16 + r) * 64 + nt * 16] = (_Float16)v;
            }
}

// ---------------------------------------------------------------------------
// conv3: 64 -> 10 (padded to 16), implicit GEMM, no ReLU, f32 output [H*W][10].
// ---------------------------------------------------------------------------
__global__ __launch_bounds__(256)
void conv3_wmma_kernel(const _Float16* __restrict__ fin,
                       const _Float16* __restrict__ bp,
                       const float* __restrict__ bias,
                       float* __restrict__ fout) {
    const int wave   = (blockIdx.x * blockDim.x + threadIdx.x) >> 5;
    const int lane   = threadIdx.x & 31;
    const int laneHi = lane >> 4;
    const int m      = lane & 15;
    const int iy     = wave >> 3;
    const int x0     = (wave & 7) << 6;

    const _Float16* baseA[4];
#pragma unroll
    for (int t = 0; t < 4; ++t)
        baseA[t] = fin + ((size_t)iy * PADW + (x0 + t * 16 + m)) * 64
                       + (laneHi ? 8 : 0);
    const _Float16* baseB = bp + (size_t)lane * 16;

    const float b0 = (m < 10) ? bias[m] : 0.f;
    v8f acc[4];
#pragma unroll
    for (int t = 0; t < 4; ++t)
#pragma unroll
        for (int r = 0; r < 8; ++r) acc[t][r] = b0;

#pragma unroll
    for (int kstep = 0; kstep < 18; ++kstep) {
        const int tap  = kstep >> 1;
        const int ty   = tap / 3, tx = tap % 3;
        const int aoff = (ty * PADW + tx) * 64 + ((kstep & 1) << 5);

        v16h a[4];
#pragma unroll
        for (int t = 0; t < 4; ++t) {
            half8 lo = *(const half8*)(baseA[t] + aoff);
            half8 hi = *(const half8*)(baseA[t] + aoff + 16);
#pragma unroll
            for (int e = 0; e < 8; ++e) { a[t][e] = lo[e]; a[t][e + 8] = hi[e]; }
        }
        const v16h b = *(const v16h*)(baseB + (size_t)kstep * 512);
#pragma unroll
        for (int t = 0; t < 4; ++t)
            acc[t] = __builtin_amdgcn_wmma_f32_16x16x32_f16(
                false, a[t], false, b, (short)0, acc[t], false, false);
    }

    if (m < 10) {
        float* obase = fout + ((size_t)iy * WW + (x0 + 8 * laneHi)) * 10 + m;
#pragma unroll
        for (int t = 0; t < 4; ++t)
#pragma unroll
            for (int r = 0; r < 8; ++r)
                obase[(size_t)(t * 16 + r) * 10] = acc[t][r];
    }
}

// ---------------------------------------------------------------------------
// gather pixels + scatter-sum into superpixels (f32 atomics, L2-resident)
// ---------------------------------------------------------------------------
__global__ __launch_bounds__(256)
void scatter_sum_kernel(const float* __restrict__ feats,
                        const int* __restrict__ pix_rc,
                        const int* __restrict__ seg,
                        float* __restrict__ sums,
                        float* __restrict__ cnt) {
    int p = blockIdx.x * blockDim.x + threadIdx.x;
    if (p >= PPIX) return;
    const int r = pix_rc[2 * p + 0];
    const int c = pix_rc[2 * p + 1];
    const int s = seg[p];
    atomicAdd(&cnt[s], 1.0f);
    const float* f = feats + ((size_t)r * WW + c) * 10;
    float* o = sums + (size_t)s * 10;
#pragma unroll
    for (int ch = 0; ch < 10; ++ch) atomicAdd(&o[ch], f[ch]);
}

__global__ void finalize_kernel(const float* __restrict__ sums,
                                const float* __restrict__ cnt,
                                float* __restrict__ out) {
    int i = blockIdx.x * blockDim.x + threadIdx.x;
    if (i >= NSP * 10) return;
    float c = cnt[i / 10];
    out[i] = sums[i] / fmaxf(c, 1.0f);
}

// ---------------------------------------------------------------------------
extern "C" void kernel_launch(void* const* d_in, const int* in_sizes, int n_in,
                              void* d_out, int out_size, void* d_ws, size_t ws_size,
                              hipStream_t stream) {
    const float* raw    = (const float*)d_in[0];
    const float* w1     = (const float*)d_in[1];
    const float* b1     = (const float*)d_in[2];
    const float* w2     = (const float*)d_in[3];
    const float* b2     = (const float*)d_in[4];
    const float* w3     = (const float*)d_in[5];
    const float* b3     = (const float*)d_in[6];
    const int*   pix_rc = (const int*)d_in[7];
    const int*   seg    = (const int*)d_in[8];
    float*       out    = (float*)d_out;

    char* ws = (char*)d_ws;
    const size_t F = (size_t)PADW * PADW * 64 * sizeof(_Float16);  // 33,817,088 B
    _Float16* feats1 = (_Float16*)(ws);
    _Float16* feats2 = (_Float16*)(ws + F);
    _Float16* bp2    = (_Float16*)(ws + 2 * F);
    _Float16* bp3    = (_Float16*)(ws + 2 * F + 73728);
    float*    feats3 = (float*)   (ws + 2 * F + 73728 + 18432);
    float*    sums   = (float*)   (ws + 2 * F + 73728 + 18432 + (size_t)PPIX * 10 * 4);
    float*    cnt    = sums + (size_t)NSP * 10;

    // zero padded feature maps (borders must be 0) and the reduction buffers
    zero16_kernel<<<4096, 256, 0, stream>>>((f4*)ws, (2 * F) / 16);
    zero16_kernel<<<16, 256, 0, stream>>>((f4*)sums, (size_t)(NSP * 10 + NSP) * 4 / 16);

    conv1_relu_kernel<<<PPIX / 256, 256, 0, stream>>>(raw, w1, b1, feats1);
    pack_w2_kernel<<<(18 * 4 * 32 + 255) / 256, 256, 0, stream>>>(w2, bp2);
    pack_w3_kernel<<<(18 * 32 + 255) / 256, 256, 0, stream>>>(w3, bp3);

    // 4096 waves each: 512 rows x 8 x-tiles of 64 pixels -> 512 blocks x 8 waves
    conv2_wmma_kernel<<<512, 256, 0, stream>>>(feats1, bp2, b2, feats2);
    conv3_wmma_kernel<<<512, 256, 0, stream>>>(feats2, bp3, b3, feats3);

    scatter_sum_kernel<<<PPIX / 256, 256, 0, stream>>>(feats3, pix_rc, seg, sums, cnt);
    finalize_kernel<<<(NSP * 10 + 255) / 256, 256, 0, stream>>>(sums, cnt, out);
}
